// spGKHDA_85847806313252
// MI455X (gfx1250) — matrix operand order repaired and checked by reading, not compiled, not run on hardware
//
#include <hip/hip_runtime.h>
#include <math.h>

typedef __attribute__((ext_vector_type(16))) _Float16 v16h;
typedef __attribute__((ext_vector_type(8)))  _Float16 v8h;
typedef __attribute__((ext_vector_type(8)))  float    v8f;

#define N_NODES 100000
#define N_EDGES 1600000
#define NFEAT   128
#define NHID    32
#define NHEADS  8
#define NCLASS  40
#define CDIM    (NHEADS * NHID)   // 256
#define C2PAD   48                // NCLASS padded to 3 WMMA tiles
#define BETA    0.1f
#define ALPHA_L 0.2f

static inline int cdiv_ll(long long a, int b) { return (int)((a + b - 1) / b); }

// ---- monotonic float <-> uint map for atomic float max --------------------
__device__ __forceinline__ unsigned fmap_u(float f) {
  unsigned b = __float_as_uint(f);
  return (b & 0x80000000u) ? ~b : (b | 0x80000000u);
}
__device__ __forceinline__ float funmap_u(unsigned u) {
  return (u & 0x80000000u) ? __uint_as_float(u ^ 0x80000000u)
                           : __uint_as_float(~u);
}
__device__ __forceinline__ float elu_f(float v) {
  return v > 0.0f ? v : (expf(v) - 1.0f);
}

// ---- conversions ----------------------------------------------------------
__global__ void k_cvt_f16(const float* __restrict__ in, _Float16* __restrict__ out,
                          long long n) {
  long long i = blockIdx.x * (long long)blockDim.x + threadIdx.x;
  if (i < n) out[i] = (_Float16)in[i];
}

// Pack W[h][k][d] (8,128,32) into WMMA B-fragment order:
//   Bp[((ct*KBLK + kb)*32 + lane)*16 + ii]
//   ii = 2j+p  ->  K = kb*32 + 2j + 16*(lane>>4) + p,  col = ct*16 + (lane&15)
__global__ void k_pack_W(const float* __restrict__ W, _Float16* __restrict__ Bp) {
  const int KBLK = NFEAT / 32;          // 4
  int i = blockIdx.x * blockDim.x + threadIdx.x;
  if (i >= NFEAT * CDIM) return;
  int ii   = i & 15;
  int lane = (i >> 4) & 31;
  int rest = i >> 9;
  int kb   = rest % KBLK;
  int ct   = rest / KBLK;
  int j = ii >> 1, p = ii & 1;
  int K   = kb * 32 + 2 * j + 16 * (lane >> 4) + p;
  int col = ct * 16 + (lane & 15);
  int h = col / NHID, d = col % NHID;
  Bp[i] = (_Float16)W[(size_t)h * NFEAT * NHID + (size_t)K * NHID + d];
}

// Pack W_out[k][c] (256,40) into B-fragment order, zero-pad cols 40..47
__global__ void k_pack_Wout(const float* __restrict__ W, _Float16* __restrict__ Bp) {
  const int KBLK = CDIM / 32;           // 8
  int i = blockIdx.x * blockDim.x + threadIdx.x;
  if (i >= CDIM * C2PAD) return;
  int ii   = i & 15;
  int lane = (i >> 4) & 31;
  int rest = i >> 9;
  int kb   = rest % KBLK;
  int ct   = rest / KBLK;
  int j = ii >> 1, p = ii & 1;
  int K   = kb * 32 + 2 * j + 16 * (lane >> 4) + p;
  int col = ct * 16 + (lane & 15);
  Bp[i] = (col < NCLASS) ? (_Float16)W[(size_t)K * NCLASS + col] : (_Float16)0.0f;
}

// ---- WMMA GEMM: D[M,Creal] = A[M,K] * Bpacked, one 16x16 tile per wave -----
// A row-major f16 [M,K]; B pre-packed per-fragment (see k_pack_*).
template <int K, int CPAD>
__global__ void k_gemm_f16_wmma(const _Float16* __restrict__ A,
                                const _Float16* __restrict__ Bp,
                                float* __restrict__ D, int M, int Creal) {
  constexpr int KBLK   = K / 32;
  constexpr int CTILES = CPAD / 16;
  int wave   = (int)((blockIdx.x * (long long)blockDim.x + threadIdx.x) >> 5);
  int lane   = threadIdx.x & 31;
  int mtiles = (M + 15) >> 4;
  int mtile  = wave / CTILES;
  int ctile  = wave - mtile * CTILES;
  if (mtile >= mtiles) return;

  int half = lane >> 4;        // 0: lanes 0-15, 1: lanes 16-31
  int idx  = lane & 15;
  int row  = mtile * 16 + idx;
  int col  = ctile * 16 + idx;
  const _Float16* arow = A + (size_t)row * K;
  const _Float16* bl   = Bp + (((size_t)ctile * KBLK) * 32 + lane) * 16;

  v8f acc = {};
#pragma unroll
  for (int kb = 0; kb < KBLK; ++kb) {
    // A fragment: two contiguous 8-half chunks per lane (b128 loads)
    v8h a0 = *(const v8h*)(arow + kb * 32 + 16 * half);
    v8h a1 = *(const v8h*)(arow + kb * 32 + 16 * half + 8);
    // B fragment: contiguous 16-half packed record per lane (2x b128)
    v8h b0 = *(const v8h*)(bl + (size_t)kb * 512);
    v8h b1 = *(const v8h*)(bl + (size_t)kb * 512 + 8);
    v16h a, b;
#pragma unroll
    for (int q = 0; q < 8; ++q) {
      a[q] = a0[q]; a[q + 8] = a1[q];
      b[q] = b0[q]; b[q + 8] = b1[q];
    }
    acc = __builtin_amdgcn_wmma_f32_16x16x32_f16(false, a, false, b,
                                                 (short)0, acc, false, false);
  }
  if (col < Creal) {
#pragma unroll
    for (int r = 0; r < 8; ++r) {
      int m = mtile * 16 + r + 8 * half;  // C/D layout: VGPR r -> M = r + 8*half
      D[(size_t)m * Creal + col] = acc[r];
    }
  }
}

// ---- s/t projections: out[h*N+n] = dot(feat[n, h*Df : h*Df+Df], a[h*Df:]) --
__global__ void k_st_dot(const float* __restrict__ feat, const float* __restrict__ avec,
                         float* __restrict__ out, int heads, int Df, int fstride) {
  long long tid = blockIdx.x * (long long)blockDim.x + threadIdx.x;
  if (tid >= (long long)heads * N_NODES) return;
  int n = (int)(tid % N_NODES);
  int h = (int)(tid / N_NODES);
  const float4* f = (const float4*)(feat + (size_t)n * fstride + (size_t)h * Df);
  const float4* a = (const float4*)(avec + (size_t)h * Df);
  float s = 0.0f;
  for (int q = 0; q < Df / 4; ++q) {
    float4 fv = f[q], av = a[q];
    s += fv.x * av.x + fv.y * av.y + fv.z * av.z + fv.w * av.w;
  }
  out[(size_t)h * N_NODES + n] = s;
}

// ---- edge logits + atomic segment max -------------------------------------
__global__ void k_edge_logit(const float* __restrict__ s, const float* __restrict__ t,
                             const int* __restrict__ src, const int* __restrict__ dst,
                             float* __restrict__ ew, unsigned* __restrict__ mmax,
                             int heads) {
  long long tid = blockIdx.x * (long long)blockDim.x + threadIdx.x;
  if (tid >= (long long)heads * N_EDGES) return;
  int e = (int)(tid % N_EDGES);
  int h = (int)(tid / N_EDGES);
  int u = src[e], v = dst[e];
  float x = s[(size_t)h * N_NODES + u] + t[(size_t)h * N_NODES + v];
  x = x > 0.0f ? x : ALPHA_L * x;                 // leaky_relu
  ew[tid] = x;
  atomicMax(&mmax[(size_t)h * N_NODES + v], fmap_u(x));
}

__global__ void k_edge_exp(const int* __restrict__ dst, float* __restrict__ ew,
                           const unsigned* __restrict__ mmax, float* __restrict__ denom,
                           int heads) {
  long long tid = blockIdx.x * (long long)blockDim.x + threadIdx.x;
  if (tid >= (long long)heads * N_EDGES) return;
  int e = (int)(tid % N_EDGES);
  int h = (int)(tid / N_EDGES);
  size_t di = (size_t)h * N_NODES + dst[e];
  float p = expf(ew[tid] - funmap_u(mmax[di]));
  ew[tid] = p;
  atomicAdd(&denom[di], p);
}

__global__ void k_edge_norm(const int* __restrict__ dst, float* __restrict__ ew,
                            const float* __restrict__ denom, int heads) {
  long long tid = blockIdx.x * (long long)blockDim.x + threadIdx.x;
  if (tid >= (long long)heads * N_EDGES) return;
  int e = (int)(tid % N_EDGES);
  int h = (int)(tid / N_EDGES);
  ew[tid] = ew[tid] / (denom[(size_t)h * N_NODES + dst[e]] + 1e-16f);
}

// ---- weighted scatter: msg[dst, h*Df:+Df] += w * feat[src, h*Df:+Df] ------
__global__ void k_scatter(const float* __restrict__ wgt, const int* __restrict__ src,
                          const int* __restrict__ dst, const float* __restrict__ feat,
                          float* __restrict__ msg, int heads, int Df, int fstride) {
  long long tid = blockIdx.x * (long long)blockDim.x + threadIdx.x;
  if (tid >= (long long)heads * N_EDGES) return;
  int e = (int)(tid % N_EDGES);
  int h = (int)(tid / N_EDGES);
  float w = wgt[tid];
  const float4* fs = (const float4*)(feat + (size_t)src[e] * fstride + (size_t)h * Df);
  float* md = msg + (size_t)dst[e] * fstride + (size_t)h * Df;
  for (int q = 0; q < Df / 4; ++q) {
    float4 v = fs[q];
    atomicAdd(md + 4 * q + 0, w * v.x);
    atomicAdd(md + 4 * q + 1, w * v.y);
    atomicAdd(md + 4 * q + 2, w * v.z);
    atomicAdd(md + 4 * q + 3, w * v.w);
  }
}

__global__ void k_update(const float* __restrict__ msg, const float* __restrict__ h0,
                         float* __restrict__ feat, long long n) {
  long long i = blockIdx.x * (long long)blockDim.x + threadIdx.x;
  if (i < n) feat[i] = (1.0f - BETA) * msg[i] + BETA * h0[i];
}

__global__ void k_elu_f16(const float* __restrict__ in, _Float16* __restrict__ out,
                          long long n) {
  long long i = blockIdx.x * (long long)blockDim.x + threadIdx.x;
  if (i < n) out[i] = (_Float16)elu_f(in[i]);
}

__global__ void k_elu_logsoftmax(const float* __restrict__ in, float* __restrict__ out) {
  int n = blockIdx.x * blockDim.x + threadIdx.x;
  if (n >= N_NODES) return;
  float v[NCLASS];
  float m = -INFINITY;
  for (int c = 0; c < NCLASS; ++c) {
    v[c] = elu_f(in[(size_t)n * NCLASS + c]);
    m = fmaxf(m, v[c]);
  }
  float s = 0.0f;
  for (int c = 0; c < NCLASS; ++c) s += expf(v[c] - m);
  float lse = m + logf(s);
  for (int c = 0; c < NCLASS; ++c) out[(size_t)n * NCLASS + c] = v[c] - lse;
}

// ---------------------------------------------------------------------------
extern "C" void kernel_launch(void* const* d_in, const int* in_sizes, int n_in,
                              void* d_out, int out_size, void* d_ws, size_t ws_size,
                              hipStream_t stream) {
  const float* x     = (const float*)d_in[0];
  const int*   ei    = (const int*)d_in[1];
  const float* W     = (const float*)d_in[2];
  const float* a_src = (const float*)d_in[3];
  const float* a_dst = (const float*)d_in[4];
  const float* W_out = (const float*)d_in[5];
  const float* a_so  = (const float*)d_in[6];
  const float* a_do  = (const float*)d_in[7];
  const int* src = ei;
  const int* dst = ei + N_EDGES;
  float* out = (float*)d_out;

  // ---- carve workspace ----
  char*  base = (char*)d_ws;
  size_t off  = 0;
  auto alloc = [&](size_t bytes) -> char* {
    char* r = base + off;
    off = (off + bytes + 255) & ~(size_t)255;
    return r;
  };
  _Float16* xh  = (_Float16*)alloc((size_t)N_NODES * NFEAT * 2);
  _Float16* Wp  = (_Float16*)alloc((size_t)NFEAT * CDIM * 2);
  _Float16* Wop = (_Float16*)alloc((size_t)CDIM * C2PAD * 2);
  float*    h0  = (float*)alloc((size_t)N_NODES * CDIM * 4);
  float*    feat= (float*)alloc((size_t)N_NODES * CDIM * 4);   // layer-2 reuses
  float*    msg = (float*)alloc((size_t)N_NODES * CDIM * 4);   // layer-2 reuses
  _Float16* h16 = (_Float16*)alloc((size_t)N_NODES * CDIM * 2);
  float*    ew  = (float*)alloc((size_t)NHEADS * N_EDGES * 4); // layer-2 reuses
  float*    sb  = (float*)alloc((size_t)NHEADS * N_NODES * 4);
  float*    tb  = (float*)alloc((size_t)NHEADS * N_NODES * 4);
  unsigned* mm  = (unsigned*)alloc((size_t)NHEADS * N_NODES * 4);
  float*    dn  = (float*)alloc((size_t)NHEADS * N_NODES * 4);
  float*    out0= (float*)alloc((size_t)N_NODES * NCLASS * 4);

  const int TPB = 256;
  const long long NC = (long long)N_NODES * CDIM;
  const long long HE = (long long)NHEADS * N_EDGES;

  // ---- precision conversions / weight packing ----
  k_cvt_f16  <<<cdiv_ll((long long)N_NODES * NFEAT, TPB), TPB, 0, stream>>>(
      x, xh, (long long)N_NODES * NFEAT);
  k_pack_W   <<<cdiv_ll(NFEAT * CDIM, TPB), TPB, 0, stream>>>(W, Wp);
  k_pack_Wout<<<cdiv_ll(CDIM * C2PAD, TPB), TPB, 0, stream>>>(W_out, Wop);

  // ---- GEMM1: h0 = x @ Wcat  (100000x128 @ 128x256) ----
  {
    long long waves = (long long)(N_NODES / 16) * (CDIM / 16);
    k_gemm_f16_wmma<NFEAT, CDIM><<<cdiv_ll(waves * 32, TPB), TPB, 0, stream>>>(
        xh, Wp, h0, N_NODES, CDIM);
  }

  // ---- layer 1 attention (8 heads) ----
  k_st_dot<<<cdiv_ll((long long)NHEADS * N_NODES, TPB), TPB, 0, stream>>>(
      h0, a_src, sb, NHEADS, NHID, CDIM);
  k_st_dot<<<cdiv_ll((long long)NHEADS * N_NODES, TPB), TPB, 0, stream>>>(
      h0, a_dst, tb, NHEADS, NHID, CDIM);
  hipMemsetAsync(mm, 0, (size_t)NHEADS * N_NODES * 4, stream);
  hipMemsetAsync(dn, 0, (size_t)NHEADS * N_NODES * 4, stream);
  k_edge_logit<<<cdiv_ll(HE, TPB), TPB, 0, stream>>>(sb, tb, src, dst, ew, mm, NHEADS);
  k_edge_exp  <<<cdiv_ll(HE, TPB), TPB, 0, stream>>>(dst, ew, mm, dn, NHEADS);
  k_edge_norm <<<cdiv_ll(HE, TPB), TPB, 0, stream>>>(dst, ew, dn, NHEADS);

  // ---- layer 1: two diffusion hops ----
  hipMemsetAsync(msg, 0, (size_t)NC * 4, stream);
  k_scatter<<<cdiv_ll(HE, TPB), TPB, 0, stream>>>(ew, src, dst, h0, msg, NHEADS, NHID, CDIM);
  k_update <<<cdiv_ll(NC, TPB), TPB, 0, stream>>>(msg, h0, feat, NC);
  hipMemsetAsync(msg, 0, (size_t)NC * 4, stream);
  k_scatter<<<cdiv_ll(HE, TPB), TPB, 0, stream>>>(ew, src, dst, feat, msg, NHEADS, NHID, CDIM);
  k_update <<<cdiv_ll(NC, TPB), TPB, 0, stream>>>(msg, h0, feat, NC);
  k_elu_f16<<<cdiv_ll(NC, TPB), TPB, 0, stream>>>(feat, h16, NC);

  // ---- GEMM2: out0 = h @ W_out  (100000x256 @ 256x40, padded to 48) ----
  {
    long long waves = (long long)(N_NODES / 16) * (C2PAD / 16);
    k_gemm_f16_wmma<CDIM, C2PAD><<<cdiv_ll(waves * 32, TPB), TPB, 0, stream>>>(
        h16, Wop, out0, N_NODES, NCLASS);
  }

  // ---- layer 2 attention (1 "head", 40 features) ----
  const long long N2 = (long long)N_NODES * NCLASS;
  k_st_dot<<<cdiv_ll((long long)N_NODES, TPB), TPB, 0, stream>>>(
      out0, a_so, sb, 1, NCLASS, NCLASS);
  k_st_dot<<<cdiv_ll((long long)N_NODES, TPB), TPB, 0, stream>>>(
      out0, a_do, tb, 1, NCLASS, NCLASS);
  hipMemsetAsync(mm, 0, (size_t)N_NODES * 4, stream);
  hipMemsetAsync(dn, 0, (size_t)N_NODES * 4, stream);
  k_edge_logit<<<cdiv_ll(N_EDGES, TPB), TPB, 0, stream>>>(sb, tb, src, dst, ew, mm, 1);
  k_edge_exp  <<<cdiv_ll(N_EDGES, TPB), TPB, 0, stream>>>(dst, ew, mm, dn, 1);
  k_edge_norm <<<cdiv_ll(N_EDGES, TPB), TPB, 0, stream>>>(dst, ew, dn, 1);

  hipMemsetAsync(msg, 0, (size_t)N2 * 4, stream);
  k_scatter<<<cdiv_ll(N_EDGES, TPB), TPB, 0, stream>>>(ew, src, dst, out0, msg, 1, NCLASS, NCLASS);
  k_update <<<cdiv_ll(N2, TPB), TPB, 0, stream>>>(msg, out0, feat, N2);
  hipMemsetAsync(msg, 0, (size_t)N2 * 4, stream);
  k_scatter<<<cdiv_ll(N_EDGES, TPB), TPB, 0, stream>>>(ew, src, dst, feat, msg, 1, NCLASS, NCLASS);
  k_update <<<cdiv_ll(N2, TPB), TPB, 0, stream>>>(msg, out0, feat, N2);

  // ---- final elu + log_softmax ----
  k_elu_logsoftmax<<<cdiv_ll((long long)N_NODES, TPB), TPB, 0, stream>>>(feat, out);
}
